// RoPEAttention_20813411516617
// MI455X (gfx1250) — compile-verified, matrix operand order checked by reading, not run on hardware
//
#include <hip/hip_runtime.h>

// ---------------------------------------------------------------------------
// 3D-RoPE attention for MI455X (gfx1250, wave32, WMMA f32<-f16 16x16x32).
//   B=2, S=2048, D=768, H=8, Hd=96.
// Pipeline:
//   0) one streaming f32->f16 conversion pass for X / W_qkv / W_out
//   1) QKV GEMM (32x32 tiles/wave, pure-f16 b128 loads) + fused 3D RoPE
//   2) flash attention, 32 Q rows (2 tiles) per wave: 24 WMMA per 32-key
//      chunk against 12 fragment loads (halves L2 K/V traffic)
//   3) output projection (32x32 tiles/wave), f32 result to d_out
// ---------------------------------------------------------------------------

typedef _Float16 v16h __attribute__((ext_vector_type(16)));
typedef _Float16 v8h  __attribute__((ext_vector_type(8)));
typedef _Float16 v4h  __attribute__((ext_vector_type(4)));
typedef float    v8f  __attribute__((ext_vector_type(8)));
typedef float    v4f  __attribute__((ext_vector_type(4)));

#define WMMA(a, b, c) \
  __builtin_amdgcn_wmma_f32_16x16x32_f16(false, (a), false, (b), (short)0, (c), false, false)

#define NB      2
#define NS      2048
#define ND      768
#define NH      8
#define NHD     96
#define RSQRT96 0.10206207261596575f
#define LOG1E4_16 0.5756462732485114f   // ln(10000)/16

// A-fragment (16x32 f16, MxK): lane holds row M = lane%16; halves are two
// contiguous 8-element runs at K = kb..kb+7 and kb+16..kb+23, where kb already
// includes the (lane&16 ? 8 : 0) half-wave shift.  16B-aligned b128 loads.
__device__ __forceinline__ v16h load_fragA_f16(const _Float16* rowbase, int kb) {
  v16h f;
  v8h lo = *(const v8h*)(rowbase + kb);
  v8h hi = *(const v8h*)(rowbase + kb + 16);
#pragma unroll
  for (int i = 0; i < 8; ++i) { f[i] = lo[i]; f[i + 8] = hi[i]; }
  return f;
}

// ---------------------------------------------------------------------------
// Kernel 0: streaming f32 -> f16 conversion (b128 in, b64 out).
// ---------------------------------------------------------------------------
__global__ __launch_bounds__(256) void cvt_f16_kernel(
    const float* __restrict__ src, _Float16* __restrict__ dst, int n) {
  const int i = (blockIdx.x * 256 + threadIdx.x) * 4;
  if (i < n) {
    v4f s = *(const v4f*)(src + i);
    v4h d;
#pragma unroll
    for (int j = 0; j < 4; ++j) d[j] = (_Float16)s[j];
    *(v4h*)(dst + i) = d;
  }
}

// ---------------------------------------------------------------------------
// Kernel 1: QKV projection (X[4096,768] x Wqkv[2304,768]^T), 32x32 tile/wave,
// fused 3D RoPE.  q -> [B,H,S,96] (pre-scaled 1/sqrt(96)), k -> [B,H,S,96],
// v -> [B,H,96,S] (transposed for PV B-fragments).
// grid: (72/8 = 9, 128);  wave w handles tile_n = bx*8+w, tile_m = by.
// ---------------------------------------------------------------------------
__global__ __launch_bounds__(256) void qkv_rope_kernel(
    const _Float16* __restrict__ xh, const float* __restrict__ coords,
    const _Float16* __restrict__ wqh,
    _Float16* __restrict__ qb, _Float16* __restrict__ kb,
    _Float16* __restrict__ vtb) {
  const int lane   = threadIdx.x & 31;
  const int tile_n = blockIdx.x * 8 + (threadIdx.x >> 5);  // 0..71
  const int tile_m = blockIdx.y;                           // 0..127
  const int l16  = lane & 15;
  const int koff = (lane & 16) ? 8 : 0;
  const int rhi  = (lane & 16) ? 8 : 0;

  const _Float16* arow0 = xh  + (size_t)(tile_m * 32 + l16) * ND;
  const _Float16* arow1 = arow0 + 16 * ND;
  const _Float16* brow0 = wqh + (size_t)(tile_n * 32 + l16) * ND;
  const _Float16* brow1 = brow0 + 16 * ND;

  v8f acc[2][2];
#pragma unroll
  for (int i = 0; i < 2; ++i)
#pragma unroll
    for (int j = 0; j < 2; ++j) acc[i][j] = (v8f){};

  for (int kt = 0; kt < 24; ++kt) {
    const int kb = kt * 32 + koff;
    __builtin_prefetch(arow0 + kb + 128, 0, 3);
    __builtin_prefetch(brow0 + kb + 128, 0, 3);
    v16h a0 = load_fragA_f16(arow0, kb);
    v16h a1 = load_fragA_f16(arow1, kb);
    v16h b0 = load_fragA_f16(brow0, kb);
    v16h b1 = load_fragA_f16(brow1, kb);
    acc[0][0] = WMMA(a0, b0, acc[0][0]);
    acc[0][1] = WMMA(a0, b1, acc[0][1]);
    acc[1][0] = WMMA(a1, b0, acc[1][0]);
    acc[1][1] = WMMA(a1, b1, acc[1][1]);
  }

  // ---- epilogue: each 16-col subtile maps to one (q|k|v, head, axis) ----
#pragma unroll
  for (int cn = 0; cn < 2; ++cn) {
    const int nbase = tile_n * 32 + cn * 16;
    const int which = nbase / ND;             // 0=q 1=k 2=v (wave-uniform)
    const int rem   = nbase % ND;
    const int head  = rem / NHD;              // subtile never straddles a head
    const int d     = (rem % NHD) + l16;      // per-lane dim within head

    if (which < 2) {                          // q or k : apply RoPE
      const int   j    = d >> 1;
      const int   axis = j >> 4;              // one axis per 16-col subtile
      const int   fidx = j & 15;
      const float freq = __expf(-(float)fidx * LOG1E4_16);
      _Float16*   outb = (which == 0) ? qb : kb;
      const float scl  = (which == 0) ? RSQRT96 : 1.0f;
#pragma unroll
      for (int cm = 0; cm < 2; ++cm) {
#pragma unroll
        for (int r = 0; r < 8; ++r) {
          const int m  = tile_m * 32 + cm * 16 + r + rhi;  // token = b*S+s
          const int bi = m >> 11, s = m & (NS - 1);
          const float th = coords[m * 3 + axis] * freq;
          float sn, cs; __sincosf(th, &sn, &cs);
          const float own = acc[cm][cn][r];
          const float oth = __shfl_xor(own, 1, 32);        // rotate partner
          float res = (d & 1) ? (oth * sn + own * cs)      // odd : x1*s + x2*c
                              : (own * cs - oth * sn);     // even: x1*c - x2*s
          outb[((size_t)(bi * NH + head) * NS + s) * NHD + d] =
              (_Float16)(res * scl);
        }
      }
    } else {                                  // v : store transposed [B,H,96,S]
#pragma unroll
      for (int cm = 0; cm < 2; ++cm) {
#pragma unroll
        for (int r = 0; r < 8; ++r) {
          const int m  = tile_m * 32 + cm * 16 + r + rhi;
          const int bi = m >> 11, s = m & (NS - 1);
          vtb[((size_t)(bi * NH + head) * NHD + d) * NS + s] =
              (_Float16)acc[cm][cn][r];
        }
      }
    }
  }
}

// ---------------------------------------------------------------------------
// Kernel 2: flash attention.  One wave owns TWO 16-row Q tiles (32 Q rows) of
// one (b,h): per 32-key chunk, 6 K-frag loads feed 12 QK^T WMMA and 6 V-frag
// loads feed 12 P*V WMMA.  P goes C-layout -> A-layout via per-wave LDS
// (DS ops are in-order within a wave, no barrier needed).
// grid: (64/8 = 8, B*H = 16).
// ---------------------------------------------------------------------------
__global__ __launch_bounds__(256) void attn_kernel(
    const _Float16* __restrict__ qb, const _Float16* __restrict__ kb,
    const _Float16* __restrict__ vtb, _Float16* __restrict__ ob) {
  __shared__ _Float16 lds_p[8][2 * 16 * 32];  // 2KB per wave (two P tiles)

  const int lane  = threadIdx.x & 31;
  const int wv    = threadIdx.x >> 5;
  const int qpair = blockIdx.x * 8 + wv;      // 0..63 (pairs of q-tiles)
  const int bh    = blockIdx.y;               // b*8+h
  const int l16   = lane & 15;
  const int koff  = (lane & 16) ? 8 : 0;
  const int rhi   = (lane & 16) ? 8 : 0;

  const _Float16* kbase = kb  + (size_t)bh * NS * NHD;
  const _Float16* vbase = vtb + (size_t)bh * NHD * NS;
  _Float16* myl = lds_p[wv];

  v16h qf[2][3];
#pragma unroll
  for (int qq = 0; qq < 2; ++qq) {
    const _Float16* qrow =
        qb + ((size_t)bh * NS + qpair * 32 + qq * 16 + l16) * NHD;
#pragma unroll
    for (int kc = 0; kc < 3; ++kc)
      qf[qq][kc] = load_fragA_f16(qrow, kc * 32 + koff);
  }

  v8f oacc[2][6];
#pragma unroll
  for (int qq = 0; qq < 2; ++qq)
#pragma unroll
    for (int dt = 0; dt < 6; ++dt) oacc[qq][dt] = (v8f){};
  float M[2][8], L[2][8], alpha[2][8];
#pragma unroll
  for (int qq = 0; qq < 2; ++qq)
#pragma unroll
    for (int r = 0; r < 8; ++r) { M[qq][r] = -__builtin_inff(); L[qq][r] = 0.0f; }

  for (int t0 = 0; t0 < NS; t0 += 32) {
    // ---- S = Q * K^T : 6 K-frag loads -> 12 WMMA ----
    v8f sc[2][2];
#pragma unroll
    for (int qq = 0; qq < 2; ++qq)
#pragma unroll
      for (int nt = 0; nt < 2; ++nt) sc[qq][nt] = (v8f){};
    const _Float16* krow0 = kbase + (size_t)(t0 + l16) * NHD;
    const _Float16* krow1 = krow0 + 16 * NHD;
#pragma unroll
    for (int kc = 0; kc < 3; ++kc) {
      v16h b0 = load_fragA_f16(krow0, kc * 32 + koff);
      v16h b1 = load_fragA_f16(krow1, kc * 32 + koff);
      sc[0][0] = WMMA(qf[0][kc], b0, sc[0][0]);
      sc[0][1] = WMMA(qf[0][kc], b1, sc[0][1]);
      sc[1][0] = WMMA(qf[1][kc], b0, sc[1][0]);
      sc[1][1] = WMMA(qf[1][kc], b1, sc[1][1]);
    }

    // ---- online softmax per q-tile: row stats via intra-half shuffles ----
#pragma unroll
    for (int qq = 0; qq < 2; ++qq) {
#pragma unroll
      for (int r = 0; r < 8; ++r) {
        float mx = fmaxf(sc[qq][0][r], sc[qq][1][r]);
        mx = fmaxf(mx, __shfl_xor(mx, 1, 32));
        mx = fmaxf(mx, __shfl_xor(mx, 2, 32));
        mx = fmaxf(mx, __shfl_xor(mx, 4, 32));
        mx = fmaxf(mx, __shfl_xor(mx, 8, 32));
        const float Mn = fmaxf(M[qq][r], mx);
        alpha[qq][r] = __expf(M[qq][r] - Mn);
        const float p0 = __expf(sc[qq][0][r] - Mn);
        const float p1 = __expf(sc[qq][1][r] - Mn);
        float rs = p0 + p1;
        rs += __shfl_xor(rs, 1, 32);
        rs += __shfl_xor(rs, 2, 32);
        rs += __shfl_xor(rs, 4, 32);
        rs += __shfl_xor(rs, 8, 32);
        L[qq][r] = L[qq][r] * alpha[qq][r] + rs;
        M[qq][r] = Mn;
        // stage P (C-layout) into LDS for the A-layout reload
        const int row = qq * 512 + (r + rhi) * 32 + l16;
        myl[row]      = (_Float16)p0;
        myl[row + 16] = (_Float16)p1;
      }
#pragma unroll
      for (int dt = 0; dt < 6; ++dt)
#pragma unroll
        for (int r = 0; r < 8; ++r) oacc[qq][dt][r] *= alpha[qq][r];
    }

    // ---- O += P * V : 6 V-frag loads -> 12 WMMA ----
    const v16h pf0 = load_fragA_f16(myl + l16 * 32, koff);
    const v16h pf1 = load_fragA_f16(myl + 512 + l16 * 32, koff);
#pragma unroll
    for (int dt = 0; dt < 6; ++dt) {
      const _Float16* vrow = vbase + (size_t)(dt * 16 + l16) * NS + t0;
      v16h vf = load_fragA_f16(vrow, koff);
      oacc[0][dt] = WMMA(pf0, vf, oacc[0][dt]);
      oacc[1][dt] = WMMA(pf1, vf, oacc[1][dt]);
    }
  }

  // ---- normalize + write O as f16 [B,S,768] for the projection GEMM ----
  const int b = bh >> 3, h = bh & 7;
#pragma unroll
  for (int qq = 0; qq < 2; ++qq) {
#pragma unroll
    for (int r = 0; r < 8; ++r) {
      const float inv = 1.0f / L[qq][r];
      const int s = qpair * 32 + qq * 16 + r + rhi;
      _Float16* orow = ob + ((size_t)(b * NS + s)) * ND + h * NHD;
#pragma unroll
      for (int dt = 0; dt < 6; ++dt)
        orow[dt * 16 + l16] = (_Float16)(oacc[qq][dt][r] * inv);
    }
  }
}

// ---------------------------------------------------------------------------
// Kernel 3: output projection O[4096,768] x Wout[768,768]^T -> f32 d_out.
// 32x32 tile/wave.  grid: (24/8 = 3, 128).
// ---------------------------------------------------------------------------
__global__ __launch_bounds__(256) void proj_kernel(
    const _Float16* __restrict__ ob, const _Float16* __restrict__ woh,
    float* __restrict__ out) {
  const int lane   = threadIdx.x & 31;
  const int tile_n = blockIdx.x * 8 + (threadIdx.x >> 5);  // 0..23
  const int tile_m = blockIdx.y;                           // 0..127
  const int l16  = lane & 15;
  const int koff = (lane & 16) ? 8 : 0;
  const int rhi  = (lane & 16) ? 8 : 0;

  const _Float16* arow0 = ob  + (size_t)(tile_m * 32 + l16) * ND;
  const _Float16* arow1 = arow0 + 16 * ND;
  const _Float16* brow0 = woh + (size_t)(tile_n * 32 + l16) * ND;
  const _Float16* brow1 = brow0 + 16 * ND;

  v8f acc[2][2];
#pragma unroll
  for (int i = 0; i < 2; ++i)
#pragma unroll
    for (int j = 0; j < 2; ++j) acc[i][j] = (v8f){};

  for (int kt = 0; kt < 24; ++kt) {
    const int kb = kt * 32 + koff;
    __builtin_prefetch(brow0 + kb + 128, 0, 3);
    v16h a0 = load_fragA_f16(arow0, kb);
    v16h a1 = load_fragA_f16(arow1, kb);
    v16h b0 = load_fragA_f16(brow0, kb);
    v16h b1 = load_fragA_f16(brow1, kb);
    acc[0][0] = WMMA(a0, b0, acc[0][0]);
    acc[0][1] = WMMA(a0, b1, acc[0][1]);
    acc[1][0] = WMMA(a1, b0, acc[1][0]);
    acc[1][1] = WMMA(a1, b1, acc[1][1]);
  }

#pragma unroll
  for (int cm = 0; cm < 2; ++cm)
#pragma unroll
    for (int cn = 0; cn < 2; ++cn)
#pragma unroll
      for (int r = 0; r < 8; ++r)
        out[(size_t)(tile_m * 32 + cm * 16 + r + rhi) * ND +
            tile_n * 32 + cn * 16 + l16] = acc[cm][cn][r];
}

// ---------------------------------------------------------------------------
extern "C" void kernel_launch(void* const* d_in, const int* in_sizes, int n_in,
                              void* d_out, int out_size, void* d_ws, size_t ws_size,
                              hipStream_t stream) {
  const float* x      = (const float*)d_in[0];   // [2,2048,768]
  const float* coords = (const float*)d_in[1];   // [2,2048,3]
  const float* Wqkv   = (const float*)d_in[2];   // [2304,768]
  const float* Wout   = (const float*)d_in[3];   // [768,768]
  float*       out    = (float*)d_out;           // [2,2048,768]

  const int XN  = NB * NS * ND;                  // 3,145,728
  const int WQN = 3 * ND * ND;                   // 1,769,472
  const int WON = ND * ND;                       //   589,824
  const size_t QN = (size_t)NB * NH * NS * NHD;  // 3,145,728

  _Float16* xh  = (_Float16*)d_ws;
  _Float16* wqh = xh  + XN;
  _Float16* woh = wqh + WQN;
  _Float16* q   = woh + WON;
  _Float16* k   = q + QN;
  _Float16* vt  = k + QN;
  _Float16* o   = vt + QN;                       // [B,S,768] f16

  // 0) one-time f32 -> f16 conversions (all L2-resident afterwards)
  cvt_f16_kernel<<<XN  / 1024, 256, 0, stream>>>(x,    xh,  XN);
  cvt_f16_kernel<<<WQN / 1024, 256, 0, stream>>>(Wqkv, wqh, WQN);
  cvt_f16_kernel<<<WON / 1024, 256, 0, stream>>>(Wout, woh, WON);

  // 1) QKV + RoPE: 128 x 72 32x32-tiles, 8 waves/block
  qkv_rope_kernel<<<dim3(9, 128), 256, 0, stream>>>(xh, coords, wqh, q, k, vt);
  // 2) attention: 64 q-tile-pairs x 16 (b,h)
  attn_kernel<<<dim3(8, 16), 256, 0, stream>>>(q, k, vt, o);
  // 3) projection: 128 x 24 32x32-tiles
  proj_kernel<<<dim3(3, 128), 256, 0, stream>>>(o, woh, out);
}
